// MoET5Block_86698209837681
// MI455X (gfx1250) — compile-verified, matrix operand order checked by reading, not run on hardware
//
#include <hip/hip_runtime.h>
#include <hip/hip_bf16.h>

// ---- problem constants ----
#define Bb   8
#define Nn   1024
#define Dd   768
#define Hh   12
#define DKV  64
#define DFF  3072
#define Ee   16
#define CAP  128
#define HD   (Hh*DKV)     // 768
#define QLD  (3*HD)       // fused qkv row stride = 2304

// Try the CDNA5 async global->LDS path (ASYNCcnt-tracked, no VGPR round trip).
#define TRY_ASYNC 1

typedef __attribute__((ext_vector_type(16))) __bf16 v16bf;
typedef __attribute__((ext_vector_type(8)))  float  v8f;

union FragBF { v16bf v; unsigned int u[8]; };
union AccU   { v8f v; float f[8]; };
union U4     { uint4 q; unsigned short s[8]; };

__device__ __forceinline__ unsigned short f2bf(float f) {
  unsigned int u = __float_as_uint(f);
  u += 0x7FFFu + ((u >> 16) & 1u);          // round-to-nearest-even
  return (unsigned short)(u >> 16);
}
__device__ __forceinline__ float bf2f(unsigned short h) {
  return __uint_as_float(((unsigned int)h) << 16);
}

__device__ __forceinline__ v8f wmma_bf16(v16bf a, v16bf b, v8f c) {
  return __builtin_amdgcn_wmma_f32_16x16x32_bf16(false, a, false, b, (short)0, c, false, false);
}

// ---- CDNA5 async global->LDS (inline asm; VDST = LDS byte address VGPR) ----
#if TRY_ASYNC
__device__ __forceinline__ void async_g2l_b128(const void* gptr, void* lptr) {
  unsigned lo = (unsigned)(unsigned long long)lptr;   // low 32 bits = wave-relative LDS offset
  asm volatile("global_load_async_to_lds_b128 %0, %1, off"
               :: "v"(lo), "v"((unsigned long long)gptr) : "memory");
}
__device__ __forceinline__ void wait_async0() {
  asm volatile("s_wait_asynccnt 0" ::: "memory");
}
#endif

// ---- 16-lane butterfly reductions: DPP (VALU) instead of ds_bpermute ----
#if defined(__has_builtin)
#if __has_builtin(__builtin_amdgcn_mov_dpp)
#define USE_DPP 1
#endif
#endif

#ifdef USE_DPP
template<int CTRL>
__device__ __forceinline__ float dppf(float x) {
  return __uint_as_float((unsigned)__builtin_amdgcn_mov_dpp(
      (int)__float_as_uint(x), CTRL, 0xF, 0xF, true));
}
__device__ __forceinline__ float red16_max(float x) {
  x = fmaxf(x, dppf<0xB1>(x));    // quad_perm(1,0,3,2)  : xor 1
  x = fmaxf(x, dppf<0x4E>(x));    // quad_perm(2,3,0,1)  : xor 2
  x = fmaxf(x, dppf<0x141>(x));   // row_half_mirror     : other quad of 8
  x = fmaxf(x, dppf<0x140>(x));   // row_mirror          : other 8 of 16
  return x;
}
__device__ __forceinline__ float red16_sum(float x) {
  x += dppf<0xB1>(x);
  x += dppf<0x4E>(x);
  x += dppf<0x141>(x);
  x += dppf<0x140>(x);
  return x;
}
#else
__device__ __forceinline__ float red16_max(float x) {
  x = fmaxf(x, __shfl_xor(x, 1)); x = fmaxf(x, __shfl_xor(x, 2));
  x = fmaxf(x, __shfl_xor(x, 4)); x = fmaxf(x, __shfl_xor(x, 8));
  return x;
}
__device__ __forceinline__ float red16_sum(float x) {
  x += __shfl_xor(x, 1); x += __shfl_xor(x, 2);
  x += __shfl_xor(x, 4); x += __shfl_xor(x, 8);
  return x;
}
#endif

// Load a 16x32 bf16 fragment (A-layout per CDNA5 ISA 7.12.2) from LDS.
__device__ __forceinline__ v16bf ld_frag(const unsigned short* base, int ld) {
  int lane = threadIdx.x & 31;
  int r = lane & 15, half = lane >> 4;
  const unsigned short* p = base + r * ld;
  FragBF f;
#pragma unroll
  for (int v = 0; v < 8; ++v) {
    int k0 = ((v & 4) << 2) + half * 8 + (v & 3) * 2;
    f.u[v] = *(const unsigned int*)(p + k0);
  }
  return f.v;
}

// =====================================================================
// Batched bf16 GEMM: C[M,N] = A[M,K] @ B[K,N] (row-major), 128x128x32
// tiles, 8 waves (4 M x 2 N of 64), double-buffered LDS pipeline with
// async global->LDS for A and in-register prefetch for transposed B.
// EPI: 0 f32 | 1 f32+residual | 2 bf16 | 3 bf16+relu
// =====================================================================
template<int EPI>
__global__ void __launch_bounds__(256)
gemm_kernel(const unsigned short* __restrict__ A, long long sA,
            const unsigned short* __restrict__ Bm, long long sB,
            void* __restrict__ Cv, long long sC,
            const float* __restrict__ R, long long sR,
            int K, int lda, int ldb, int ldc)
{
  __shared__ unsigned short Al[2][128][40];
  __shared__ unsigned short Bl[2][128][40];   // transposed: Bl[buf][n][k]

  const int z = blockIdx.z;
  A  += (long long)z * sA;
  Bm += (long long)z * sB;

  const int tid = threadIdx.x;
  const int wave = tid >> 5, lane = tid & 31;
  const int ln = lane & 15, half = lane >> 4;
  const int m0 = blockIdx.y * 128, n0 = blockIdx.x * 128;
  const int wm = (wave & 3) * 32;
  const int wn = (wave >> 2) * 64;

  // per-thread staging coords
  const int ar = tid >> 1, ac = (tid & 1) * 16;      // A: 16 elems
  const int bk = tid >> 3, bn = (tid & 7) * 16;      // B: 16 elems (transpose)

  AccU acc[2][4];
#pragma unroll
  for (int i = 0; i < 2; ++i)
#pragma unroll
    for (int j = 0; j < 4; ++j)
#pragma unroll
      for (int r = 0; r < 8; ++r) acc[i][j].f[r] = 0.f;

  auto stageA = [&](int buf, int k0) {
    const unsigned short* g = A + (long long)(m0 + ar) * lda + k0 + ac;
#if TRY_ASYNC
    async_g2l_b128(g,     &Al[buf][ar][ac]);
    async_g2l_b128(g + 8, &Al[buf][ar][ac + 8]);
#else
    *(uint4*)&Al[buf][ar][ac]     = *(const uint4*)g;
    *(uint4*)&Al[buf][ar][ac + 8] = *(const uint4*)(g + 8);
#endif
  };
  auto loadB = [&](int k0, U4& lo, U4& hi) {
    const unsigned short* g = Bm + (long long)(k0 + bk) * ldb + n0 + bn;
    lo.q = *(const uint4*)g;
    hi.q = *(const uint4*)(g + 8);
  };
  auto storeB = [&](int buf, const U4& lo, const U4& hi) {
#pragma unroll
    for (int i = 0; i < 8; ++i) Bl[buf][bn + i][bk]     = lo.s[i];
#pragma unroll
    for (int i = 0; i < 8; ++i) Bl[buf][bn + 8 + i][bk] = hi.s[i];
  };

  // prologue: stage tile 0 into buffer 0
  {
    U4 lo, hi;
    stageA(0, 0);
    loadB(0, lo, hi);
    storeB(0, lo, hi);
#if TRY_ASYNC
    wait_async0();
#endif
    __syncthreads();
  }

  int buf = 0;
  for (int k0 = 0; k0 < K; k0 += 32) {
    const bool has_next = (k0 + 32) < K;
    U4 nlo, nhi;
    if (has_next) {               // prefetch tile t+1: async A + B into regs
      stageA(buf ^ 1, k0 + 32);
      loadB(k0 + 32, nlo, nhi);
    }
    // compute tile t (8 WMMAs)
    v16bf a0 = ld_frag(&Al[buf][wm][0], 40);
    v16bf a1 = ld_frag(&Al[buf][wm + 16][0], 40);
#pragma unroll
    for (int j = 0; j < 4; ++j) {
      v16bf bf = ld_frag(&Bl[buf][wn + j * 16][0], 40);
      acc[0][j].v = wmma_bf16(a0, bf, acc[0][j].v);
      acc[1][j].v = wmma_bf16(a1, bf, acc[1][j].v);
    }
    if (has_next) storeB(buf ^ 1, nlo, nhi);
#if TRY_ASYNC
    if (has_next) wait_async0();
#endif
    __syncthreads();
    buf ^= 1;
  }

  // epilogue (C/D layout: m = r + 8*half, n = lane&15)
  float* Cf = (float*)Cv + (long long)z * sC;
  unsigned short* Cb = (unsigned short*)Cv + (long long)z * sC;
  const float* Rz = R ? R + (long long)z * sR : nullptr;
#pragma unroll
  for (int i = 0; i < 2; ++i)
#pragma unroll
    for (int j = 0; j < 4; ++j)
#pragma unroll
      for (int r = 0; r < 8; ++r) {
        int mC = m0 + wm + i * 16 + half * 8 + r;
        int nC = n0 + wn + j * 16 + ln;
        long long off = (long long)mC * ldc + nC;
        float val = acc[i][j].f[r];
        if (EPI == 0) Cf[off] = val;
        if (EPI == 1) Cf[off] = val + Rz[off];
        if (EPI == 2) Cb[off] = f2bf(val);
        if (EPI == 3) Cb[off] = f2bf(fmaxf(val, 0.f));
      }
}

// =====================================================================
// Flash attention, T5 style (no scale, additive mask bias), fused QKV
// input buffer [T, 2304]. 8 waves x 16 query rows, 32-key tiles.
// =====================================================================
__global__ void __launch_bounds__(256)
flash_attn_kernel(const unsigned short* __restrict__ qkv,
                  const unsigned char* __restrict__ mask,
                  unsigned short* __restrict__ o)
{
  __shared__ unsigned short Kl[32][72];        // [key][d]
  __shared__ unsigned short Vt[64][40];        // [d][key] (transposed V)
  __shared__ unsigned short Pl[8][16][40];     // per-wave P tile [m][key]
  __shared__ float bias_sh[32];

  const int tid = threadIdx.x, wave = tid >> 5, lane = tid & 31;
  const int ln = lane & 15, half = lane >> 4;
  const int bh = blockIdx.y, b = bh / Hh, h = bh % Hh;
  const int q0 = blockIdx.x * 128 + wave * 16;

  const unsigned short* qbase = qkv + h * DKV;             // q at col 0
  const unsigned short* kbase = qkv + HD + h * DKV;        // k at col 768
  const unsigned short* vbase = qkv + 2 * HD + h * DKV;    // v at col 1536

  // Q fragments straight from global (A-layout; K pairs contiguous)
  FragBF qa0, qa1;
  {
    const unsigned short* qp = qbase + ((size_t)(b * Nn + q0 + ln)) * QLD;
#pragma unroll
    for (int vv = 0; vv < 8; ++vv) {
      int kk = ((vv & 4) << 2) + half * 8 + (vv & 3) * 2;
      qa0.u[vv] = *(const unsigned int*)(qp + kk);
      qa1.u[vv] = *(const unsigned int*)(qp + 32 + kk);
    }
  }

  AccU oacc[4];
#pragma unroll
  for (int t = 0; t < 4; ++t)
#pragma unroll
    for (int r = 0; r < 8; ++r) oacc[t].f[r] = 0.f;
  float mrow[8], lrow[8];
#pragma unroll
  for (int r = 0; r < 8; ++r) { mrow[r] = -1e30f; lrow[r] = 0.f; }

  for (int kb = 0; kb < Nn; kb += 32) {
    __syncthreads();
    const int key = tid >> 3, d = (tid & 7) * 8;
    {   // stage K tile [key][d] (async direct to LDS)
      const unsigned short* g = kbase + ((size_t)(b * Nn + kb + key)) * QLD + d;
#if TRY_ASYNC
      async_g2l_b128(g, &Kl[key][d]);
#else
      *(uint4*)&Kl[key][d] = *(const uint4*)g;
#endif
    }
    {   // stage V tile transposed Vt[d][key]
      const unsigned short* g = vbase + ((size_t)(b * Nn + kb + key)) * QLD + d;
      U4 t8; t8.q = *(const uint4*)g;
#pragma unroll
      for (int i = 0; i < 8; ++i) Vt[d + i][key] = t8.s[i];
    }
    if (tid < 32) bias_sh[tid] = mask[b * Nn + kb + tid] ? 0.f : -1e9f;
#if TRY_ASYNC
    wait_async0();
#endif
    __syncthreads();

    // S = Q @ K^T  (two 16x16 key tiles, K-dim 64 -> 2 WMMAs each)
    AccU s0, s1;
#pragma unroll
    for (int r = 0; r < 8; ++r) { s0.f[r] = 0.f; s1.f[r] = 0.f; }
    s0.v = wmma_bf16(qa0.v, ld_frag(&Kl[0][0], 72),  s0.v);
    s0.v = wmma_bf16(qa1.v, ld_frag(&Kl[0][32], 72), s0.v);
    s1.v = wmma_bf16(qa0.v, ld_frag(&Kl[16][0], 72),  s1.v);
    s1.v = wmma_bf16(qa1.v, ld_frag(&Kl[16][32], 72), s1.v);

    float bi0 = bias_sh[ln], bi1 = bias_sh[16 + ln];
#pragma unroll
    for (int r = 0; r < 8; ++r) {
      float x0 = s0.f[r] + bi0, x1 = s1.f[r] + bi1;
      float mx = red16_max(fmaxf(x0, x1));            // DPP butterfly (VALU)
      float mnew = fmaxf(mrow[r], mx);
      float p0 = __expf(x0 - mnew), p1 = __expf(x1 - mnew);
      float rs = red16_sum(p0 + p1);
      float corr = __expf(mrow[r] - mnew);
      lrow[r] = lrow[r] * corr + rs;
      mrow[r] = mnew;
#pragma unroll
      for (int t = 0; t < 4; ++t) oacc[t].f[r] *= corr;
      int m = r + 8 * half;
      Pl[wave][m][ln]      = f2bf(p0);
      Pl[wave][m][16 + ln] = f2bf(p1);
    }
    // O += P @ V  (wave-local P; same-wave DS ordering compiler-waited)
    v16bf pf = ld_frag(&Pl[wave][0][0], 40);
#pragma unroll
    for (int t = 0; t < 4; ++t) {
      v16bf vf = ld_frag(&Vt[t * 16][0], 40);
      oacc[t].v = wmma_bf16(pf, vf, oacc[t].v);
    }
  }

#pragma unroll
  for (int r = 0; r < 8; ++r) {
    float inv = 1.f / lrow[r];
    int row = q0 + r + 8 * half;
#pragma unroll
    for (int t = 0; t < 4; ++t)
      o[((size_t)(b * Nn + row)) * HD + h * DKV + t * 16 + ln] = f2bf(oacc[t].f[r] * inv);
  }
}

// ---------------------------------------------------------------------
__global__ void cast_kernel(const float* __restrict__ in,
                            unsigned short* __restrict__ out, size_t n)
{
  size_t i = (size_t)blockIdx.x * blockDim.x + threadIdx.x;
  size_t stride = (size_t)gridDim.x * blockDim.x;
  for (; i < n; i += stride) out[i] = f2bf(in[i]);
}

// strided cast: pack [rows, cols] f32 into bf16 rows of stride ldo
__global__ void cast_stride_kernel(const float* __restrict__ in,
                                   unsigned short* __restrict__ out,
                                   int cols, int ldo, size_t total)
{
  size_t i = (size_t)blockIdx.x * blockDim.x + threadIdx.x;
  size_t stride = (size_t)gridDim.x * blockDim.x;
  for (; i < total; i += stride) {
    size_t r = i / cols, c = i - r * cols;
    out[r * ldo + c] = f2bf(in[i]);
  }
}

// T5 RMSNorm: one block per row of 768
__global__ void __launch_bounds__(256)
rmsnorm_kernel(const float* __restrict__ x, const float* __restrict__ w,
               unsigned short* __restrict__ out)
{
  int row = blockIdx.x;
  const float* xr = x + (size_t)row * Dd;
  float ss = 0.f;
  for (int i = threadIdx.x; i < Dd; i += 256) { float v = xr[i]; ss += v * v; }
  ss = red16_sum(ss);
  ss += __shfl_xor(ss, 16);
  __shared__ float sh[8];
  if ((threadIdx.x & 31) == 0) sh[threadIdx.x >> 5] = ss;
  __syncthreads();
  float tot = 0.f;
#pragma unroll
  for (int i = 0; i < 8; ++i) tot += sh[i];
  float scale = rsqrtf(tot / (float)Dd + 1e-6f);
  for (int i = threadIdx.x; i < Dd; i += 256)
    out[(size_t)row * Dd + i] = f2bf(xr[i] * scale * w[i]);
}

// top-2 gating: one wave per token; lanes 0..15 = experts
__global__ void __launch_bounds__(256)
gating_kernel(const unsigned short* __restrict__ x, const unsigned short* __restrict__ gw,
              float* __restrict__ rawbuf, int* __restrict__ idx1o, int* __restrict__ idx2o,
              float* __restrict__ g1o, float* __restrict__ g2o)
{
  int wave = threadIdx.x >> 5, lane = threadIdx.x & 31;
  int tok = blockIdx.x * 8 + wave;
  float acc = 0.f;
  if (lane < 16) {
    const unsigned short* xr = x + (size_t)tok * Dd;
    for (int d = 0; d < Dd; ++d) acc += bf2f(xr[d]) * bf2f(gw[d * Ee + lane]);
  }
  float mx = red16_max(acc);
  float ex = __expf(acc - mx);
  float sm = red16_sum(ex);
  float raw = ex / sm;
  if (lane < 16) rawbuf[(size_t)tok * Ee + lane] = raw;
  // argmax #1 (tie -> lowest index, matches jnp.argmax)
  float v1 = raw; int id1 = lane & 15;
#pragma unroll
  for (int off = 1; off < 16; off <<= 1) {
    float ov = __shfl_xor(v1, off); int oi = __shfl_xor(id1, off);
    if (ov > v1 || (ov == v1 && oi < id1)) { v1 = ov; id1 = oi; }
  }
  float raw2 = ((lane & 15) == id1) ? -1.f : raw;
  float v2 = raw2; int id2 = lane & 15;
#pragma unroll
  for (int off = 1; off < 16; off <<= 1) {
    float ov = __shfl_xor(v2, off); int oi = __shfl_xor(id2, off);
    if (ov > v2 || (ov == v2 && oi < id2)) { v2 = ov; id2 = oi; }
  }
  float den = v1 + v2 + 1e-9f;
  if (lane == 0) {
    idx1o[tok] = id1; idx2o[tok] = id2;
    g1o[tok] = v1 / den; g2o[tok] = v2 / den;
  }
}

// capacity scan (exact exclusive-cumsum semantics); one wave per batch
__global__ void __launch_bounds__(32)
scan_kernel(const int* __restrict__ idx1, const int* __restrict__ idx2,
            const float* __restrict__ g1, const float* __restrict__ g2,
            int* __restrict__ disp, int* __restrict__ slot1, int* __restrict__ slot2,
            float* __restrict__ w1g, float* __restrict__ w2g)
{
  int b = blockIdx.x, lane = threadIdx.x;
  int cnt = 0;
  for (int n = 0; n < Nn; ++n) {
    int t = b * Nn + n;
    int i1 = idx1[t];
    int c = __shfl(cnt, i1);
    if (lane == i1) cnt++;
    if (lane == 0) {
      bool ok = c < CAP;
      int s = (i1 * Bb + b) * CAP + c;
      slot1[t] = ok ? s : -1;
      w1g[t]   = ok ? g1[t] : 0.f;
      if (ok) disp[s] = n;
    }
  }
  int m1c = min(cnt, CAP);
  int cnt2 = 0;
  for (int n = 0; n < Nn; ++n) {
    int t = b * Nn + n;
    int i2 = idx2[t];
    int c = __shfl(cnt2, i2) + __shfl(m1c, i2);
    if (lane == i2) cnt2++;
    if (lane == 0) {
      bool ok = c < CAP;
      int s = (i2 * Bb + b) * CAP + c;
      slot2[t] = ok ? s : -1;
      w2g[t]   = ok ? g2[t] : 0.f;
      if (ok) disp[s] = n;
    }
  }
}

// gather tokens into expert slots: one block per (e,b,c) row
__global__ void __launch_bounds__(192)
gather_kernel(const int* __restrict__ disp, const unsigned short* __restrict__ x2,
              unsigned short* __restrict__ xin)
{
  int row = blockIdx.x;
  int tok = disp[row];
  int b = (row / CAP) & (Bb - 1);
  uint2* dst = (uint2*)(xin + (size_t)row * Dd) + threadIdx.x;
  if (tok >= 0) {
    const uint2* src = (const uint2*)(x2 + ((size_t)(b * Nn + tok)) * Dd) + threadIdx.x;
    *dst = *src;
  } else {
    *dst = make_uint2(0u, 0u);
  }
}

// out = h + g1*xo[slot1] + g2*xo[slot2]; one block per token
__global__ void __launch_bounds__(192)
combine_kernel(const float* __restrict__ h, const float* __restrict__ xo,
               const int* __restrict__ slot1, const int* __restrict__ slot2,
               const float* __restrict__ w1g, const float* __restrict__ w2g,
               float* __restrict__ out)
{
  int t = blockIdx.x, j = threadIdx.x;
  float4 v = ((const float4*)(h + (size_t)t * Dd))[j];
  int s1 = slot1[t];
  if (s1 >= 0) {
    float g = w1g[t];
    float4 e = ((const float4*)(xo + (size_t)s1 * Dd))[j];
    v.x += g * e.x; v.y += g * e.y; v.z += g * e.z; v.w += g * e.w;
  }
  int s2 = slot2[t];
  if (s2 >= 0) {
    float g = w2g[t];
    float4 e = ((const float4*)(xo + (size_t)s2 * Dd))[j];
    v.x += g * e.x; v.y += g * e.y; v.z += g * e.z; v.w += g * e.w;
  }
  ((float4*)(out + (size_t)t * Dd))[j] = v;
}

// deterministic aux loss: thread i = (b,e) pair; serial sums, tree combine
__global__ void __launch_bounds__(128)
loss_kernel(const float* __restrict__ raw, const int* __restrict__ idx1,
            float* __restrict__ outv)
{
  int i = threadIdx.x, b = i >> 4, e = i & 15;
  float s = 0.f; int c = 0;
  for (int n = 0; n < Nn; ++n) {
    int t = b * Nn + n;
    s += raw[(size_t)t * Ee + e];
    c += (idx1[t] == e) ? 1 : 0;
  }
  float v = (s / (float)Nn) * ((float)c / (float)Nn);
  __shared__ float sh[128];
  sh[i] = v; __syncthreads();
  for (int st = 64; st > 0; st >>= 1) { if (i < st) sh[i] += sh[i + st]; __syncthreads(); }
  if (i == 0) outv[0] = sh[0] / (float)(Bb * Ee) * (float)(Ee * Ee) * 0.01f;
}

// =====================================================================
extern "C" void kernel_launch(void* const* d_in, const int* in_sizes, int n_in,
                              void* d_out, int out_size, void* d_ws, size_t ws_size,
                              hipStream_t stream)
{
  (void)in_sizes; (void)n_in; (void)out_size; (void)ws_size;
  const float* hidden = (const float*)d_in[0];
  const unsigned char* mask = (const unsigned char*)d_in[1];
  const float* ln0  = (const float*)d_in[2];
  const float* Wq   = (const float*)d_in[3];
  const float* Wk   = (const float*)d_in[4];
  const float* Wv   = (const float*)d_in[5];
  const float* Wo   = (const float*)d_in[6];
  const float* ln1  = (const float*)d_in[7];
  const float* gw   = (const float*)d_in[8];
  const float* w1   = (const float*)d_in[9];
  const float* w2   = (const float*)d_in[10];
  float* out = (float*)d_out;

  const size_t T = (size_t)Bb * Nn;                 // 8192 tokens
  char* ws = (char*)d_ws;
  size_t cur = 0;
  auto alloc = [&](size_t bytes) -> char* {
    char* p = ws + cur;
    cur = (cur + bytes + 255) & ~(size_t)255;
    return p;
  };

  unsigned short* wqkv_b = (unsigned short*)alloc((size_t)Dd * QLD * 2);  // fused [768,2304]
  unsigned short* wo_b   = (unsigned short*)alloc((size_t)HD * Dd * 2);
  unsigned short* gw_b   = (unsigned short*)alloc((size_t)Dd * Ee * 2);
  unsigned short* w1_b   = (unsigned short*)alloc((size_t)Ee * Dd * DFF * 2);
  unsigned short* w2_b   = (unsigned short*)alloc((size_t)Ee * DFF * Dd * 2);
  unsigned short* nrm_b  = (unsigned short*)alloc(T * Dd * 2);
  unsigned short* qkv_b  = (unsigned short*)alloc(T * QLD * 2);
  unsigned short* at_b   = (unsigned short*)alloc(T * HD * 2);
  float*          h_f    = (float*)alloc(T * Dd * 4);
  unsigned short* n2_b   = (unsigned short*)alloc(T * Dd * 2);
  unsigned short* xin_b  = (unsigned short*)alloc((size_t)Ee * Bb * CAP * Dd * 2);
  unsigned short* hmid   = (unsigned short*)alloc((size_t)Ee * Bb * CAP * DFF * 2);
  float*          xo_f   = (float*)alloc((size_t)Ee * Bb * CAP * Dd * 4);
  float*          rawb   = (float*)alloc(T * Ee * 4);
  int*   idx1 = (int*)alloc(T * 4);
  int*   idx2 = (int*)alloc(T * 4);
  float* g1   = (float*)alloc(T * 4);
  float* g2   = (float*)alloc(T * 4);
  int*   sl1  = (int*)alloc(T * 4);
  int*   sl2  = (int*)alloc(T * 4);
  float* w1g  = (float*)alloc(T * 4);
  float* w2g  = (float*)alloc(T * 4);
  int*   disp = (int*)alloc((size_t)Ee * Bb * CAP * 4);

  auto cast = [&](const float* src, unsigned short* dst, size_t n) {
    int blocks = (int)((n + 2047) / 2048); if (blocks > 8192) blocks = 8192;
    cast_kernel<<<blocks, 256, 0, stream>>>(src, dst, n);
  };
  auto cast_packed = [&](const float* src, unsigned short* dst, int cols, int ldo, size_t rows) {
    size_t n = rows * (size_t)cols;
    int blocks = (int)((n + 2047) / 2048); if (blocks > 8192) blocks = 8192;
    cast_stride_kernel<<<blocks, 256, 0, stream>>>(src, dst, cols, ldo, n);
  };
  auto gemm = [&](int epi, const unsigned short* A, long long sA,
                  const unsigned short* Bm, long long sB,
                  void* C, long long sC, const float* R, long long sR,
                  int M, int Ncols, int K, int lda, int ldb, int ldc, int batch) {
    dim3 g(Ncols / 128, M / 128, batch), blk(256);
    switch (epi) {
      case 0: gemm_kernel<0><<<g, blk, 0, stream>>>(A, sA, Bm, sB, C, sC, R, sR, K, lda, ldb, ldc); break;
      case 1: gemm_kernel<1><<<g, blk, 0, stream>>>(A, sA, Bm, sB, C, sC, R, sR, K, lda, ldb, ldc); break;
      case 2: gemm_kernel<2><<<g, blk, 0, stream>>>(A, sA, Bm, sB, C, sC, R, sR, K, lda, ldb, ldc); break;
      default: gemm_kernel<3><<<g, blk, 0, stream>>>(A, sA, Bm, sB, C, sC, R, sR, K, lda, ldb, ldc); break;
    }
  };

  // 0) weight casts f32 -> bf16; Q/K/V packed into one [768, 2304] matrix
  cast_packed(Wq, wqkv_b + 0 * HD, HD, QLD, Dd);
  cast_packed(Wk, wqkv_b + 1 * HD, HD, QLD, Dd);
  cast_packed(Wv, wqkv_b + 2 * HD, HD, QLD, Dd);
  cast(Wo, wo_b, (size_t)HD * Dd);
  cast(gw, gw_b, (size_t)Dd * Ee);
  cast(w1, w1_b, (size_t)Ee * Dd * DFF);
  cast(w2, w2_b, (size_t)Ee * DFF * Dd);
  hipMemsetAsync(disp, 0xFF, (size_t)Ee * Bb * CAP * 4, stream);

  // 1) RMSNorm(hidden) -> bf16
  rmsnorm_kernel<<<(int)T, 256, 0, stream>>>(hidden, ln0, nrm_b);

  // 2) fused QKV projection (one WMMA GEMM, N=2304)
  gemm(2, nrm_b, 0, wqkv_b, 0, qkv_b, 0, nullptr, 0, (int)T, QLD, Dd, Dd, QLD, QLD, 1);

  // 3) flash attention
  flash_attn_kernel<<<dim3(Nn / 128, Bb * Hh), 256, 0, stream>>>(qkv_b, mask, at_b);

  // 4) output projection + residual -> h (f32)
  gemm(1, at_b, 0, wo_b, 0, h_f, 0, hidden, 0, (int)T, Dd, HD, HD, Dd, Dd, 1);

  // 5) RMSNorm(h) -> bf16
  rmsnorm_kernel<<<(int)T, 256, 0, stream>>>(h_f, ln1, n2_b);

  // 6) top-2 gating + capacity scan + dispatch gather
  gating_kernel<<<(int)(T / 8), 256, 0, stream>>>(n2_b, gw_b, rawb, idx1, idx2, g1, g2);
  scan_kernel<<<Bb, 32, 0, stream>>>(idx1, idx2, g1, g2, disp, sl1, sl2, w1g, w2g);
  gather_kernel<<<Ee * Bb * CAP, 192, 0, stream>>>(disp, n2_b, xin_b);

  // 7) expert FFN: batched over 16 experts (WMMA, relu fused)
  gemm(3, xin_b, (long long)Bb * CAP * Dd, w1_b, (long long)Dd * DFF,
       hmid, (long long)Bb * CAP * DFF, nullptr, 0,
       Bb * CAP, DFF, Dd, Dd, DFF, DFF, Ee);
  gemm(0, hmid, (long long)Bb * CAP * DFF, w2_b, (long long)DFF * Dd,
       xo_f, (long long)Bb * CAP * Dd, nullptr, 0,
       Bb * CAP, Dd, DFF, DFF, Dd, Dd, Ee);

  // 8) weighted combine + residual -> layer_output; aux loss
  combine_kernel<<<(int)T, 192, 0, stream>>>(h_f, xo_f, sl1, sl2, w1g, w2g, out);
  loss_kernel<<<1, 128, 0, stream>>>(rawb, idx1, out + (size_t)Bb * Nn * Dd);
}
